// RecurrentActorCritic_75943611728540
// MI455X (gfx1250) — compile-verified, matrix operand order checked by reading, not run on hardware
//
#include <hip/hip_runtime.h>

typedef unsigned short u16;
typedef __attribute__((ext_vector_type(16))) __bf16 bf16x16;
typedef __attribute__((ext_vector_type(8)))  float  f32x8;
typedef __attribute__((ext_vector_type(4)))  unsigned int u32x4;

union FragAB { bf16x16 v; uint4 q[2]; };

__device__ __forceinline__ u16 f2bf(float f) {
  unsigned u = __float_as_uint(f);
  u += 0x7FFFu + ((u >> 16) & 1u);
  return (u16)(u >> 16);
}
__device__ __forceinline__ float sigm(float x) { return 1.0f / (1.0f + __expf(-x)); }
__device__ __forceinline__ float silu_f(float x) { return x * sigm(x); }

// A fragment: lane L holds row (L&15); k = kt*32 + (L>>4)*8 + {0..7} and +16
__device__ __forceinline__ void load_a_frag(FragAB& f, const u16* s, int rowBase,
                                            int stride, int kt, int lane) {
  const u16* p = s + (rowBase + (lane & 15)) * stride + kt * 32 + ((lane >> 4) << 3);
  f.q[0] = *(const uint4*)p;
  f.q[1] = *(const uint4*)(p + 16);
}
// B fragment from packed tiles: per tile 32 lanes x 16 bf16 contiguous (1KB/tile)
__device__ __forceinline__ void load_b_frag(FragAB& f, const u16* base, int tile, int lane) {
  const u16* p = base + (size_t)tile * 512 + lane * 16;
  f.q[0] = *(const uint4*)p;
  f.q[1] = *(const uint4*)(p + 8);
}
__device__ __forceinline__ f32x8 wmma_bf16(const FragAB& a, const FragAB& b, f32x8 c) {
  return __builtin_amdgcn_wmma_f32_16x16x32_bf16(false, a.v, false, b.v, (short)0, c,
                                                 false, false);
}
__device__ __forceinline__ void unpack_bf8(uint4 q, float* o) {
  unsigned a0 = q.x, a1 = q.y, a2 = q.z, a3 = q.w;
  o[0] = __uint_as_float(a0 << 16); o[1] = __uint_as_float(a0 & 0xFFFF0000u);
  o[2] = __uint_as_float(a1 << 16); o[3] = __uint_as_float(a1 & 0xFFFF0000u);
  o[4] = __uint_as_float(a2 << 16); o[5] = __uint_as_float(a2 & 0xFFFF0000u);
  o[6] = __uint_as_float(a3 << 16); o[7] = __uint_as_float(a3 & 0xFFFF0000u);
}

// CDNA5 async DMA: copy 16B/lane from global to LDS, tracked with ASYNCcnt
__device__ __forceinline__ void async_copy_b128(void* lds_ptr, const void* gptr) {
  unsigned lds_off = (unsigned)(size_t)lds_ptr;              // LDS byte address
  unsigned long long ga = (unsigned long long)(size_t)gptr;  // 64-bit global addr
  asm volatile("global_load_async_to_lds_b128 %0, %1, off" ::"v"(lds_off), "v"(ga)
               : "memory");
}

// ---------------- weight packing: fp32 [N,K] -> bf16 WMMA-B fragment tiles ----------
__global__ __launch_bounds__(256) void pack_weight_bf16(const float* __restrict__ W,
                                                        u16* __restrict__ dst,
                                                        int N, int K) {
  int total = N * K;
  int idx = blockIdx.x * 256 + threadIdx.x;
  if (idx >= total) return;
  int NT = N >> 4;
  int tile = idx >> 9;
  int within = idx & 511;
  int lane = within >> 4;
  int e = within & 15;
  int kt = tile / NT;
  int nt = tile - kt * NT;
  int k = kt * 32 + ((e < 8) ? 0 : 16) + ((lane >> 4) << 3) + (e & 7);
  int n = nt * 16 + (lane & 15);
  dst[idx] = f2bf(W[(size_t)n * K + k]);
}

// ---------------- fused encoder: LN(silu(obs@W1^T+b1))@W2^T ... @Wih^T --------
__global__ __launch_bounds__(256) void encoder_kernel(
    const float* __restrict__ obs, const float* __restrict__ b1,
    const float* __restrict__ g1, const float* __restrict__ be1,
    const float* __restrict__ b2, const float* __restrict__ bih,
    const u16* __restrict__ gW1, const u16* __restrict__ gW2,
    const u16* __restrict__ gWih, u16* __restrict__ gGX) {
  extern __shared__ char smem[];
  u16* s_obs = (u16*)smem;                    // 128x128 bf16 (32KB)
  u16* s_x   = (u16*)(smem + 32768);          // 128x256 bf16 (64KB)
  u16* s_z   = (u16*)(smem + 98304);          // 128x192 bf16 (48KB)
  float* s_pb1  = (float*)(smem + 147456);    // 256
  float* s_pg1  = s_pb1 + 256;
  float* s_pbe1 = s_pg1 + 256;
  float* s_pb2  = s_pbe1 + 256;               // 192
  float* s_pbih = s_pb2 + 192;                // 576

  const int tid = threadIdx.x;
  const int lane = tid & 31;
  const int w = tid >> 5;
  const int slab = blockIdx.x;
  const int colb = lane & 15;

  if (tid < 256) { s_pb1[tid] = b1[tid]; s_pg1[tid] = g1[tid]; s_pbe1[tid] = be1[tid]; }
  if (tid < 192) s_pb2[tid] = b2[tid];
  for (int i = tid; i < 576; i += 256) s_pbih[i] = bih[i];

  // stage obs slab as bf16
  const float4* obs4 = (const float4*)(obs + (size_t)slab * 128 * 128);
  uint2* s_obs2 = (uint2*)s_obs;
#pragma unroll
  for (int j = 0; j < 16; ++j) {
    int v = tid + j * 256;
    float4 f = obs4[v];
    uint2 p;
    p.x = (unsigned)f2bf(f.x) | ((unsigned)f2bf(f.y) << 16);
    p.y = (unsigned)f2bf(f.z) | ((unsigned)f2bf(f.w) << 16);
    s_obs2[v] = p;
  }
  __syncthreads();

  // ---- GEMM1: [128x128] @ W1^T -> [128x256], + b1, SiLU, LayerNorm -> s_x (bf16)
  {
    f32x8 acc[16];
#pragma unroll
    for (int n = 0; n < 16; ++n) acc[n] = (f32x8){0, 0, 0, 0, 0, 0, 0, 0};
#pragma unroll
    for (int kt = 0; kt < 4; ++kt) {
      FragAB a; load_a_frag(a, s_obs, w * 16, 128, kt, lane);
      FragAB bc; load_b_frag(bc, gW1, kt * 16, lane);
#pragma unroll
      for (int n = 0; n < 16; ++n) {
        FragAB bn;
        if (n < 15) load_b_frag(bn, gW1, kt * 16 + n + 1, lane);  // pipeline next B
        acc[n] = wmma_bf16(a, bc, acc[n]);
        if (n < 15) bc = bn;
      }
    }
    float s1[8], s2[8];
#pragma unroll
    for (int i = 0; i < 8; ++i) { s1[i] = 0.f; s2[i] = 0.f; }
#pragma unroll
    for (int n = 0; n < 16; ++n) {
      float bb = s_pb1[n * 16 + colb];
#pragma unroll
      for (int i = 0; i < 8; ++i) {
        float v = silu_f(acc[n][i] + bb);
        acc[n][i] = v;
        s1[i] += v; s2[i] += v * v;
      }
    }
#pragma unroll
    for (int i = 0; i < 8; ++i) {
#pragma unroll
      for (int m = 1; m < 16; m <<= 1) {
        s1[i] += __shfl_xor(s1[i], m, 32);
        s2[i] += __shfl_xor(s2[i], m, 32);
      }
    }
    float mu[8], rstd[8];
#pragma unroll
    for (int i = 0; i < 8; ++i) {
      mu[i] = s1[i] * (1.0f / 256.0f);
      float var = s2[i] * (1.0f / 256.0f) - mu[i] * mu[i];
      rstd[i] = rsqrtf(var + 1e-5f);
    }
#pragma unroll
    for (int n = 0; n < 16; ++n) {
      int col = n * 16 + colb;
      float g = s_pg1[col], be = s_pbe1[col];
#pragma unroll
      for (int i = 0; i < 8; ++i) {
        int row = w * 16 + ((lane >> 4) << 3) + i;
        s_x[row * 256 + col] = f2bf((acc[n][i] - mu[i]) * rstd[i] * g + be);
      }
    }
  }

  // ---- GEMM2: [128x256] @ W2^T -> [128x192], + b2, SiLU -> s_z (bf16)
  {
    f32x8 acc[12];
#pragma unroll
    for (int n = 0; n < 12; ++n) acc[n] = (f32x8){0, 0, 0, 0, 0, 0, 0, 0};
#pragma unroll
    for (int kt = 0; kt < 8; ++kt) {
      FragAB a; load_a_frag(a, s_x, w * 16, 256, kt, lane);
      FragAB bc; load_b_frag(bc, gW2, kt * 12, lane);
#pragma unroll
      for (int n = 0; n < 12; ++n) {
        FragAB bn;
        if (n < 11) load_b_frag(bn, gW2, kt * 12 + n + 1, lane);
        acc[n] = wmma_bf16(a, bc, acc[n]);
        if (n < 11) bc = bn;
      }
    }
#pragma unroll
    for (int n = 0; n < 12; ++n) {
      int col = n * 16 + colb;
      float bb = s_pb2[col];
#pragma unroll
      for (int i = 0; i < 8; ++i) {
        int row = w * 16 + ((lane >> 4) << 3) + i;
        s_z[row * 192 + col] = f2bf(silu_f(acc[n][i] + bb));
      }
    }
  }

  // ---- GEMM3: [128x192] @ Wih^T + bih -> gx fragments (bf16, non-temporal) --------
  {
    FragAB af[6];
#pragma unroll
    for (int kt = 0; kt < 6; ++kt) load_a_frag(af[kt], s_z, w * 16, 192, kt, lane);
    size_t rt = (size_t)slab * 8 + w;
#pragma unroll 4
    for (int n = 0; n < 36; ++n) {
      f32x8 acc = (f32x8){0, 0, 0, 0, 0, 0, 0, 0};
      FragAB bc; load_b_frag(bc, gWih, n, lane);
#pragma unroll
      for (int kt = 0; kt < 6; ++kt) {
        FragAB bn;
        if (kt < 5) load_b_frag(bn, gWih, (kt + 1) * 36 + n, lane);
        acc = wmma_bf16(af[kt], bc, acc);
        if (kt < 5) bc = bn;
      }
      float bb = s_pbih[n * 16 + colb];
      u32x4 qo;
      qo.x = (unsigned)f2bf(acc[0] + bb) | ((unsigned)f2bf(acc[1] + bb) << 16);
      qo.y = (unsigned)f2bf(acc[2] + bb) | ((unsigned)f2bf(acc[3] + bb) << 16);
      qo.z = (unsigned)f2bf(acc[4] + bb) | ((unsigned)f2bf(acc[5] + bb) << 16);
      qo.w = (unsigned)f2bf(acc[6] + bb) | ((unsigned)f2bf(acc[7] + bb) << 16);
      __builtin_nontemporal_store(qo, (u32x4*)(gGX + (rt * 36 + n) * 256 + lane * 8));
    }
  }
}

// ---------------- persistent GRU scan + heads (Whh resident in LDS) -----------------
// LDS map (bytes):
//   0      s_whh   221184   (6*36 B-tiles, bf16)
//  221184  s_hbf     6144   (16x192 bf16 h)
//  227328  s_h      12288   (16x192 f32 h)
//  239616  s_bhh     2304
//  241920  s_wpv     4608   (Wp rows 0..4, Wv row 5)
//  246528  s_hb        32
//  246560  s_head     384
//  246944  s_start     64   (pad to 247040)
//  247040  s_gx     36864   (2 x 18432 double-buffered gx fragments)
__global__ __launch_bounds__(256) void gru_scan_kernel(
    const float* __restrict__ starts, const int* __restrict__ actions,
    const u16* __restrict__ gWhh, const u16* __restrict__ gGX,
    const float* __restrict__ bhh, const float* __restrict__ Wp,
    const float* __restrict__ bp, const float* __restrict__ Wv,
    const float* __restrict__ bv, float* __restrict__ out) {
  extern __shared__ char smem[];
  u16*   s_whh   = (u16*)smem;
  u16*   s_hbf   = (u16*)(smem + 221184);
  float* s_h     = (float*)(smem + 227328);
  float* s_bhh   = (float*)(smem + 239616);
  float* s_wpv   = (float*)(smem + 241920);
  float* s_hb    = (float*)(smem + 246528);
  float* s_head  = (float*)(smem + 246560);
  float* s_start = (float*)(smem + 246944);
  char*  s_gx    = smem + 247040;

  const int tid = threadIdx.x;
  const int lane = tid & 31;
  const int w = tid >> 5;
  const int wg = blockIdx.x;
  const int TB = 1024 * 256;
  const int nIter = (w < 4) ? 5 : 4;  // wave-uniform split of 1152 b128 transfers

  // ---- async-stage Whh (221184 B = 13824 x b128 = 54/thread) + first gx buffer ----
  {
    const char* srcW = (const char*)gWhh;
    for (int j = 0; j < 54; ++j) {
      int i = tid + j * 256;
      async_copy_b128((char*)s_whh + (size_t)i * 16, srcW + (size_t)i * 16);
    }
    const char* src0 = (const char*)(gGX + (size_t)wg * 36 * 256);
    for (int j = 0; j < nIter; ++j) {
      int i = tid + j * 256;
      async_copy_b128(s_gx + (size_t)i * 16, src0 + (size_t)i * 16);
    }
    for (int i = tid; i < 576; i += 256) s_bhh[i] = bhh[i];
    for (int i = tid; i < 5 * 192; i += 256) s_wpv[i] = Wp[i];
    if (tid < 192) s_wpv[5 * 192 + tid] = Wv[tid];
    if (tid < 5) s_hb[tid] = bp[tid];
    if (tid == 5) s_hb[5] = bv[0];
    for (int i = tid; i < 3072; i += 256) { s_h[i] = 0.f; s_hbf[i] = 0; }
    asm volatile("s_wait_asynccnt 0x0" ::: "memory");
  }
  __syncthreads();

  for (int t = 0; t < 1024; ++t) {
    if (tid < 16) s_start[tid] = starts[(size_t)t * 256 + wg * 16 + tid];
    __syncthreads();

    // episode-boundary reset
    for (int i = tid; i < 3072; i += 256) {
      float hv = s_h[i] * (1.0f - s_start[i / 192]);
      s_h[i] = hv;
      s_hbf[i] = f2bf(hv);
    }

    // kick off async copy of next step's gx into the other LDS buffer, then wait
    // for THIS step's copy (in-order async-load completion: cnt<=N_new => old done)
    if (t + 1 < 1024) {
      const char* src = (const char*)(gGX + (size_t)((t + 1) * 16 + wg) * 36 * 256);
      char* dstb = s_gx + ((t + 1) & 1) * 18432;
      for (int j = 0; j < nIter; ++j) {
        int i = tid + j * 256;
        async_copy_b128(dstb + (size_t)i * 16, src + (size_t)i * 16);
      }
      if (w < 4) asm volatile("s_wait_asynccnt 0x5" ::: "memory");
      else       asm volatile("s_wait_asynccnt 0x4" ::: "memory");
    } else {
      asm volatile("s_wait_asynccnt 0x0" ::: "memory");
    }
    __syncthreads();  // h reset done + gx[t] buffer landed (all waves)

    const u16* gxb = (const u16*)(s_gx + (t & 1) * 18432);

    FragAB af[6];
#pragma unroll
    for (int kt = 0; kt < 6; ++kt) load_a_frag(af[kt], s_hbf, 0, 192, kt, lane);

    float hnew[2][8];
#pragma unroll
    for (int gi = 0; gi < 2; ++gi) {
      int g = w + gi * 8;
      if (g < 12) {
        f32x8 ar = (f32x8){0, 0, 0, 0, 0, 0, 0, 0};
        f32x8 au = ar, an = ar;
#pragma unroll
        for (int kt = 0; kt < 6; ++kt) {
          FragAB b;
          load_b_frag(b, s_whh, kt * 36 + g, lane);       ar = wmma_bf16(af[kt], b, ar);
          load_b_frag(b, s_whh, kt * 36 + g + 12, lane);  au = wmma_bf16(af[kt], b, au);
          load_b_frag(b, s_whh, kt * 36 + g + 24, lane);  an = wmma_bf16(af[kt], b, an);
        }
        uint4 qr = *(const uint4*)(gxb + (g) * 256 + lane * 8);
        uint4 qu = *(const uint4*)(gxb + (g + 12) * 256 + lane * 8);
        uint4 qn = *(const uint4*)(gxb + (g + 24) * 256 + lane * 8);
        float gxr[8], gxu[8], gxn[8];
        unpack_bf8(qr, gxr); unpack_bf8(qu, gxu); unpack_bf8(qn, gxn);
        int col = g * 16 + (lane & 15);
        float br = s_bhh[col], bu = s_bhh[192 + col], bn = s_bhh[384 + col];
#pragma unroll
        for (int i = 0; i < 8; ++i) {
          int row = ((lane >> 4) << 3) + i;
          float hold = s_h[row * 192 + col];
          float rg = sigm(gxr[i] + ar[i] + br);
          float ug = sigm(gxu[i] + au[i] + bu);
          float ng = tanhf(gxn[i] + rg * (an[i] + bn));
          hnew[gi][i] = (1.0f - ug) * ng + ug * hold;
        }
      }
    }
    __syncthreads();  // all WMMA/gate reads of h done
#pragma unroll
    for (int gi = 0; gi < 2; ++gi) {
      int g = w + gi * 8;
      if (g < 12) {
        int col = g * 16 + (lane & 15);
#pragma unroll
        for (int i = 0; i < 8; ++i) {
          int row = ((lane >> 4) << 3) + i;
          float hv = hnew[gi][i];
          s_h[row * 192 + col] = hv;
          s_hbf[row * 192 + col] = f2bf(hv);
        }
      }
    }
    __syncthreads();  // h_new visible

    // policy/value heads
    if (tid < 96) {
      int hd = tid >> 4, row = tid & 15;
      const float* wrow = s_wpv + hd * 192;
      float acc = s_hb[hd];
      for (int c = 0; c < 192; ++c) acc += s_h[row * 192 + c] * wrow[c];
      s_head[row * 6 + hd] = acc;
    }
    __syncthreads();

    if (tid < 16) {
      int row = tid;
      float l[5], m = -1e30f;
#pragma unroll
      for (int a = 0; a < 5; ++a) { l[a] = s_head[row * 6 + a]; m = fmaxf(m, l[a]); }
      float se = 0.f;
#pragma unroll
      for (int a = 0; a < 5; ++a) se += __expf(l[a] - m);
      float logZ = m + __logf(se);
      float ent = 0.f;
#pragma unroll
      for (int a = 0; a < 5; ++a) {
        float lp = l[a] - logZ;
        ent -= __expf(lp) * lp;
      }
      size_t idx = (size_t)t * 256 + wg * 16 + row;
      int act = actions[idx];
      out[idx] = l[act] - logZ;
      out[(size_t)TB + idx] = ent;
      out[2 * (size_t)TB + idx] = s_head[row * 6 + 5];
    }
    // loop-top barrier separates head reads from next reset's writes
  }
}

extern "C" void kernel_launch(void* const* d_in, const int* in_sizes, int n_in,
                              void* d_out, int out_size, void* d_ws, size_t ws_size,
                              hipStream_t stream) {
  (void)in_sizes; (void)n_in; (void)out_size; (void)ws_size;
  const float* obs    = (const float*)d_in[0];
  const float* starts = (const float*)d_in[1];
  const int*   acts   = (const int*)d_in[2];
  const float* W1  = (const float*)d_in[3];
  const float* b1  = (const float*)d_in[4];
  const float* g1  = (const float*)d_in[5];
  const float* be1 = (const float*)d_in[6];
  const float* W2  = (const float*)d_in[7];
  const float* b2  = (const float*)d_in[8];
  const float* Wih = (const float*)d_in[9];
  const float* Whh = (const float*)d_in[10];
  const float* bih = (const float*)d_in[11];
  const float* bhh = (const float*)d_in[12];
  const float* Wp  = (const float*)d_in[13];
  const float* bp  = (const float*)d_in[14];
  const float* Wv  = (const float*)d_in[15];
  const float* bv  = (const float*)d_in[16];

  u16* ws   = (u16*)d_ws;
  u16* gW1  = ws;                  // 256*128
  u16* gW2  = gW1 + 32768;         // 192*256
  u16* gWih = gW2 + 49152;         // 576*192
  u16* gWhh = gWih + 110592;       // 576*192
  u16* gGX  = gWhh + 110592;       // 1024*256*576 bf16 fragments

  pack_weight_bf16<<<32768 / 256, 256, 0, stream>>>(W1, gW1, 256, 128);
  pack_weight_bf16<<<49152 / 256, 256, 0, stream>>>(W2, gW2, 192, 256);
  pack_weight_bf16<<<110592 / 256, 256, 0, stream>>>(Wih, gWih, 576, 192);
  pack_weight_bf16<<<110592 / 256, 256, 0, stream>>>(Whh, gWhh, 576, 192);

  encoder_kernel<<<2048, 256, 153600, stream>>>(obs, b1, g1, be1, b2, bih,
                                                gW1, gW2, gWih, gGX);
  gru_scan_kernel<<<16, 256, 283904, stream>>>(starts, acts, gWhh, gGX, bhh,
                                               Wp, bp, Wv, bv, (float*)d_out);
}
